// VLALayer_19009525252574
// MI455X (gfx1250) — matrix operand order repaired and checked by reading, not compiled, not run on hardware
//
#include <hip/hip_runtime.h>

// ---------------- CDNA5 (gfx1250, wave32) WMMA types ----------------
typedef __attribute__((ext_vector_type(16))) __bf16 bf16x16;
typedef __attribute__((ext_vector_type(8)))  float  floatx8;

#define WMMA_BF16(a, b, c) \
  __builtin_amdgcn_wmma_f32_16x16x32_bf16(false, (a), false, (b), (short)0, (c), false, false)

static constexpr int Bb = 4;
static constexpr int Tt = 2048;
static constexpr int Dd = 1024;
static constexpr int Hh = 128;
static constexpr int BT = Bb * Tt;
static constexpr int CHUNK = 64;            // attention chunk (4 M-tiles)
static constexpr int NCH = Tt / CHUNK;      // 32 chunks per batch

union FragU { bf16x16 v; uint4 q[2]; };

__device__ inline floatx8 zero8() {
  floatx8 z = {0.f, 0.f, 0.f, 0.f, 0.f, 0.f, 0.f, 0.f};
  return z;
}

// A-fragment (16x32 bf16, M x K), source row-major [row][k], ld in elements.
// Lane l: row = l%16; elements 0..7 -> K = koff..koff+7, 8..15 -> K = koff+16..+23,
// koff = 0 (lanes 0-15) or 8 (lanes 16-31).  Two 16B loads.
__device__ inline bf16x16 load_a_frag(const __bf16* base, int ld, int row, int kbase) {
  int l = threadIdx.x & 31;
  int m = l & 15;
  int koff = (l < 16) ? 0 : 8;
  const __bf16* p = base + (size_t)(row + m) * ld + kbase + koff;
  FragU f;
  f.q[0] = *reinterpret_cast<const uint4*>(p);
  f.q[1] = *reinterpret_cast<const uint4*>(p + 16);
  return f.v;
}

// B-fragment (32x16 bf16, K x N), source row-major [k][n], ld in elements.
// Lane l: col N = nbase + l%16; elements 0..7 -> K = koff..koff+7, 8..15 -> K = koff+16..+23,
// koff = 0 (lanes 0-15) or 8 (lanes 16-31).  Strided scalar loads.
__device__ inline bf16x16 load_b_frag(const __bf16* base, int ld, int kbase, int nbase) {
  int l = threadIdx.x & 31;
  int koff = (l < 16) ? 0 : 8;
  const __bf16* p = base + (size_t)(kbase + koff) * ld + nbase + (l & 15);
  bf16x16 v;
#pragma unroll
  for (int s = 0; s < 8; s++) {
    v[s] = p[(size_t)s * ld];
    v[s + 8] = p[(size_t)(s + 16) * ld];
  }
  return v;
}

// Async-stage one contiguous 64x128 bf16 chunk (16 KB) global -> LDS.
// 256 threads x 4 x b128.  Tracked by ASYNCcnt (8 issued per chunk incl. both calls).
// Per-lane LDS offset = low 32 bits of the generic pointer (LDS aperture: addr[31:0]).
__device__ inline void async_stage_chunk(const __bf16* __restrict__ gsrc, __bf16* ldst) {
  uint32_t lds0 = (uint32_t)(size_t)ldst;
  uint64_t g0 = (uint64_t)(size_t)gsrc;
  uint32_t off = (uint32_t)threadIdx.x * 16u;
#pragma unroll
  for (int it = 0; it < 4; ++it) {
    uint32_t l = lds0 + off;
    uint64_t a = g0 + off;
    asm volatile("global_load_async_to_lds_b128 %0, %1, off"
                 :: "v"(l), "v"(a) : "memory");
    off += 256u * 16u;
  }
}

// ---------------- conversion / transpose staging ----------------
__global__ void cvt_x_kernel(const float* __restrict__ x, __bf16* __restrict__ xbf, int n) {
  int i = blockIdx.x * 256 + threadIdx.x;
  if (i < n) xbf[i] = (__bf16)x[i];
}

// Wt[w][d][h] = W_w[h][d]  (bf16, transposed -> contiguous B-fragments)
__global__ void cvt_w_kernel(const float* __restrict__ Wq, const float* __restrict__ Wk,
                             const float* __restrict__ Wv, __bf16* __restrict__ Wt) {
  int w = blockIdx.y;
  int i = blockIdx.x * 256 + threadIdx.x;
  if (i >= Dd * Hh) return;
  int d = i / Hh, h = i % Hh;
  const float* W = (w == 0) ? Wq : ((w == 1) ? Wk : Wv);
  Wt[(size_t)w * Dd * Hh + i] = (__bf16)W[h * Dd + d];
}

// Wot[h][d] = Wo[d][h]
__global__ void cvt_wo_kernel(const float* __restrict__ Wo, __bf16* __restrict__ Wot) {
  int i = blockIdx.x * 256 + threadIdx.x;
  if (i >= Hh * Dd) return;
  int h = i / Dd, d = i % Dd;
  Wot[i] = (__bf16)Wo[d * Hh + h];
}

// ---------------- QKV projection GEMM (WMMA bf16) ----------------
// 1 wave per block; block computes a full 16x128 tile of Q, K, or V over K=D=1024.
// A-fragment loaded once per K-chunk and reused across all 8 N-tiles (8x reuse).
__global__ void qkv_kernel(const __bf16* __restrict__ xbf, const __bf16* __restrict__ Wt,
                           const float* __restrict__ bq, const float* __restrict__ bk,
                           const float* __restrict__ bv,
                           float* __restrict__ Qf, float* __restrict__ Kf,
                           __bf16* __restrict__ Qb, __bf16* __restrict__ Vb,
                           __bf16* __restrict__ VTb) {
  int row0 = blockIdx.x * 16;       // token tile
  int which = blockIdx.y;           // 0=Q 1=K 2=V
  const __bf16* W = Wt + (size_t)which * Dd * Hh;
  const float* bias = (which == 0) ? bq : ((which == 1) ? bk : bv);

  floatx8 acc[8];
#pragma unroll
  for (int nt = 0; nt < 8; nt++) acc[nt] = zero8();

  for (int kc = 0; kc < Dd; kc += 32) {
    bf16x16 a = load_a_frag(xbf, Dd, row0, kc);
#pragma unroll
    for (int nt = 0; nt < 8; nt++) {
      bf16x16 bb = load_b_frag(W, Hh, kc, nt * 16);
      acc[nt] = WMMA_BF16(a, bb, acc[nt]);
    }
  }

  int l = threadIdx.x & 31;
  int col = l & 15;
  int rbase = (l >> 4) * 8;
#pragma unroll
  for (int nt = 0; nt < 8; nt++) {
    float bi = bias[nt * 16 + col];
#pragma unroll
    for (int r = 0; r < 8; r++) {
      int row = row0 + rbase + r;                 // C/D layout: VGPR r -> M=r / r+8
      float v = acc[nt][r] + bi;
      if (which == 0) {
        Qf[(size_t)row * Hh + nt * 16 + col] = v;
        Qb[(size_t)row * Hh + nt * 16 + col] = (__bf16)v;
      } else if (which == 1) {
        Kf[(size_t)row * Hh + nt * 16 + col] = v;
      } else {
        Vb[(size_t)row * Hh + nt * 16 + col] = (__bf16)v;
        int b = row / Tt, t = row % Tt;           // 16-row tiles never straddle a batch
        VTb[((size_t)b * Hh + nt * 16 + col) * Tt + t] = (__bf16)v;
      }
    }
  }
}

// ---------------- Phase A: sequential Sherman-Morrison scan -> alpha ----------------
__device__ inline float block_reduce_sum(float v, float* red) {
  for (int o = 16; o > 0; o >>= 1) v += __shfl_down(v, o, 32);
  int wid = threadIdx.x >> 5;
  if ((threadIdx.x & 31) == 0) red[wid] = v;
  __syncthreads();
  float s = red[0] + red[1] + red[2] + red[3];
  __syncthreads();
  return s;
}

// 1 block per batch, 128 threads; thread j owns row j of symmetric A in registers.
// q/k for step t+1 are software-prefetched while step t computes.
__global__ void alpha_scan_kernel(const float* __restrict__ Qf, const float* __restrict__ Kf,
                                  __bf16* __restrict__ Ab, __bf16* __restrict__ ATb) {
  int b = blockIdx.x;
  int j = threadIdx.x;
  __shared__ float u_s[Hh];
  __shared__ float au_s[Hh];
  __shared__ float red[4];
  float Arow[Hh];
#pragma unroll
  for (int i = 0; i < Hh; i++) Arow[i] = (i == j) ? 1.0f : 0.0f;   // A0 = I / lambda_0

  float q = Qf[(size_t)b * Tt * Hh + j];
  float k = Kf[(size_t)b * Tt * Hh + j];
  for (int t = 0; t < Tt; t++) {
    int tn = (t + 1 < Tt) ? (t + 1) : t;
    float qn = Qf[((size_t)b * Tt + tn) * Hh + j];   // prefetch next step
    float kn = Kf[((size_t)b * Tt + tn) * Hh + j];

    float s_t = block_reduce_sum(k * q, red);
    float kk  = block_reduce_sum(k * k, red);
    float u = k * (1.0f / (sqrtf(kk) + 1e-6f));
    u_s[j] = u;
    __syncthreads();
    float au = 0.0f;
#pragma unroll
    for (int i = 0; i < Hh; i++) au += Arow[i] * u_s[i];
    float denom = 1.0f + block_reduce_sum(u * au, red);
    au_s[j] = au;
    __syncthreads();
    float rden = 1.0f / denom;
#pragma unroll
    for (int i = 0; i < Hh; i++) Arow[i] -= au * au_s[i] * rden;   // A -= (Au)(Au)^T/denom
    float alpha = s_t * au * rden;                                  // s_t * z,  z = Au/denom
    Ab[((size_t)b * Tt + t) * Hh + j] = (__bf16)alpha;
    ATb[((size_t)b * Hh + j) * Tt + t] = (__bf16)alpha;
    __syncthreads();
    q = qn;
    k = kn;
  }
}

// ---------------- Phase B: chunked causal linear attention (WMMA-dense) ----------------
// 1 block per batch, 8 waves; wave w owns output/state H-tile i in [16w,16w+16).
// Next chunk's V_c / alpha_c are async-staged to LDS (ASYNCcnt) while this chunk computes.
// Dynamic LDS layout (bytes):
//   [0,32K)    MT   : M^T bf16 mirror of running state
//   [32K,40K)  Sl   : masked intra-chunk scores, bf16
//   [40K,56K)  Vst0 | [56K,72K) Vst1 : double-buffered V chunk (64x128 bf16)
//   [72K,88K)  Ast0 | [88K,104K) Ast1 : double-buffered alpha chunk
__global__ void attn_kernel(const __bf16* __restrict__ Qb, const __bf16* __restrict__ Vb,
                            const __bf16* __restrict__ VTb, const __bf16* __restrict__ Ab,
                            const __bf16* __restrict__ ATb, __bf16* __restrict__ Ob) {
  extern __shared__ char smem[];
  __bf16* MT = (__bf16*)smem;
  __bf16* Sl = (__bf16*)(smem + 32768);
  __bf16* Vst0 = (__bf16*)(smem + 40960);
  __bf16* Vst1 = (__bf16*)(smem + 40960 + 16384);
  __bf16* Ast0 = (__bf16*)(smem + 40960 + 32768);
  __bf16* Ast1 = (__bf16*)(smem + 40960 + 49152);

  int b = blockIdx.x;
  int wid = threadIdx.x >> 5;
  int l = threadIdx.x & 31;
  int col = l & 15;
  int rbase = (l >> 4) * 8;

  const __bf16* Qbb = Qb + (size_t)b * Tt * Hh;
  const __bf16* Vbb = Vb + (size_t)b * Tt * Hh;
  const __bf16* VTbb = VTb + (size_t)b * Hh * Tt;
  const __bf16* Abb = Ab + (size_t)b * Tt * Hh;
  const __bf16* ATbb = ATb + (size_t)b * Hh * Tt;

  // kick off chunk 0 staging, then zero M^T while it flies
  async_stage_chunk(Vbb, Vst0);
  async_stage_chunk(Abb, Ast0);
  for (int i = threadIdx.x; i < Hh * Hh; i += blockDim.x) MT[i] = (__bf16)0.0f;

  floatx8 macc[8];                   // running M rows [16w,16w+16) x all 128 cols (f32)
#pragma unroll
  for (int jt = 0; jt < 8; jt++) macc[jt] = zero8();
  __syncthreads();

  for (int c = 0; c < NCH; c++) {
    int tb = c * CHUNK;
    __bf16* Vcur = (c & 1) ? Vst1 : Vst0;
    __bf16* Acur = (c & 1) ? Ast1 : Ast0;
    __bf16* Vnxt = (c & 1) ? Vst0 : Vst1;
    __bf16* Anxt = (c & 1) ? Ast0 : Ast1;

    // prefetch next chunk (clamped; redundant on the last chunk, harmless)
    int nc = (c + 1 < NCH) ? (c + 1) : c;
    async_stage_chunk(Vbb + (size_t)nc * CHUNK * Hh, Vnxt);
    async_stage_chunk(Abb + (size_t)nc * CHUNK * Hh, Anxt);

    // S = Q_c alpha_c^T  (64x64, K=H=128); 2 tiles per wave; mask s<=t; to LDS (bf16)
#pragma unroll
    for (int si = 0; si < 2; si++) {
      int id = wid * 2 + si;
      int mt = id >> 2, st = id & 3;
      floatx8 sacc = zero8();
#pragma unroll
      for (int kc = 0; kc < 4; kc++) {
        bf16x16 a = load_a_frag(Qbb, Hh, tb + mt * 16, kc * 32);
        bf16x16 bb = load_b_frag(ATbb, Tt, kc * 32, tb + st * 16);
        sacc = WMMA_BF16(a, bb, sacc);
      }
#pragma unroll
      for (int r = 0; r < 8; r++) {
        int tl = mt * 16 + rbase + r;
        int sl = st * 16 + col;
        float v = (sl <= tl) ? sacc[r] : 0.0f;   // inclusive causal mask
        Sl[tl * CHUNK + sl] = (__bf16)v;
      }
    }
    // retire this chunk's staged V/alpha (the 8 just-issued prefetch ops stay in flight)
    asm volatile("s_wait_asynccnt 0x8" ::: "memory");
    __syncthreads();

    // O_c = mask(S) V_c + Q_c M_prev^T  (wave w -> cols [16w,16w+16))
    floatx8 oacc[4];
#pragma unroll
    for (int mt = 0; mt < 4; mt++) {
      oacc[mt] = zero8();
#pragma unroll
      for (int kc = 0; kc < 2; kc++) {         // intra-chunk, K = 64 (V from LDS stage)
        bf16x16 a = load_a_frag(Sl, CHUNK, mt * 16, kc * 32);
        bf16x16 bb = load_b_frag(Vcur, Hh, kc * 32, wid * 16);
        oacc[mt] = WMMA_BF16(a, bb, oacc[mt]);
      }
#pragma unroll
      for (int kc = 0; kc < 4; kc++) {         // inter-chunk state, K = 128
        bf16x16 a = load_a_frag(Qbb, Hh, tb + mt * 16, kc * 32);
        bf16x16 bb = load_b_frag(MT, Hh, kc * 32, wid * 16);   // M_prev^T from LDS
        oacc[mt] = WMMA_BF16(a, bb, oacc[mt]);
      }
#pragma unroll
      for (int r = 0; r < 8; r++) {
        int row = tb + mt * 16 + rbase + r;
        Ob[((size_t)b * Tt + row) * Hh + wid * 16 + col] = (__bf16)oacc[mt][r];
      }
    }

    // M += V_c^T alpha_c   (wave w: rows [16w,16w+16), all 128 cols; K = 64; alpha from LDS)
#pragma unroll
    for (int kc = 0; kc < 2; kc++) {
      bf16x16 a = load_a_frag(VTbb, Tt, wid * 16, tb + kc * 32);
#pragma unroll
      for (int jt = 0; jt < 8; jt++) {
        bf16x16 bb = load_b_frag(Acur, Hh, kc * 32, jt * 16);
        macc[jt] = WMMA_BF16(a, bb, macc[jt]);
      }
    }
    __syncthreads();   // everyone done reading M_prev^T / Sl / stage buffers

    // publish updated M^T (bf16) for next chunk's O2
#pragma unroll
    for (int jt = 0; jt < 8; jt++) {
#pragma unroll
      for (int r = 0; r < 8; r++) {
        int i = wid * 16 + rbase + r;
        int jj = jt * 16 + col;
        MT[jj * Hh + i] = (__bf16)macc[jt][r];
      }
    }
    __syncthreads();
  }
}

// ---------------- Output projection GEMM (WMMA bf16) ----------------
__global__ void outproj_kernel(const __bf16* __restrict__ Ob, const __bf16* __restrict__ Wot,
                               const float* __restrict__ bo, float* __restrict__ out) {
  int row0 = blockIdx.x * 16;
  int nb0 = blockIdx.y * 32;
  floatx8 acc0 = zero8(), acc1 = zero8();
#pragma unroll
  for (int kc = 0; kc < 4; kc++) {             // K = H = 128
    bf16x16 a = load_a_frag(Ob, Hh, row0, kc * 32);
    bf16x16 b0 = load_b_frag(Wot, Dd, kc * 32, nb0);
    bf16x16 b1 = load_b_frag(Wot, Dd, kc * 32, nb0 + 16);
    acc0 = WMMA_BF16(a, b0, acc0);
    acc1 = WMMA_BF16(a, b1, acc1);
  }
  int l = threadIdx.x & 31;
  int col = l & 15;
  int rbase = (l >> 4) * 8;
  float bi0 = bo[nb0 + col], bi1 = bo[nb0 + 16 + col];
#pragma unroll
  for (int r = 0; r < 8; r++) {
    int row = row0 + rbase + r;
    out[(size_t)row * Dd + nb0 + col] = acc0[r] + bi0;
    out[(size_t)row * Dd + nb0 + 16 + col] = acc1[r] + bi1;
  }
}

// ---------------- host launcher ----------------
extern "C" void kernel_launch(void* const* d_in, const int* in_sizes, int n_in,
                              void* d_out, int out_size, void* d_ws, size_t ws_size,
                              hipStream_t stream) {
  (void)in_sizes; (void)n_in; (void)out_size; (void)ws_size;
  const float* x  = (const float*)d_in[0];
  const float* Wq = (const float*)d_in[1];
  const float* bq = (const float*)d_in[2];
  const float* Wk = (const float*)d_in[3];
  const float* bk = (const float*)d_in[4];
  const float* Wv = (const float*)d_in[5];
  const float* bv = (const float*)d_in[6];
  const float* Wo = (const float*)d_in[7];
  const float* bo = (const float*)d_in[8];
  float* out = (float*)d_out;

  char* ws = (char*)d_ws;
  size_t off = 0;
  auto carve = [&](size_t bytes) -> void* {
    void* p = ws + off;
    off += (bytes + 255) & ~(size_t)255;
    return p;
  };
  __bf16* xbf = (__bf16*)carve((size_t)BT * Dd * 2);   // 16 MB
  __bf16* Wt  = (__bf16*)carve((size_t)3 * Dd * Hh * 2);
  __bf16* Wot = (__bf16*)carve((size_t)Dd * Hh * 2);
  float*  Qf  = (float*) carve((size_t)BT * Hh * 4);
  float*  Kf  = (float*) carve((size_t)BT * Hh * 4);
  __bf16* Qb  = (__bf16*)carve((size_t)BT * Hh * 2);
  __bf16* Vb  = (__bf16*)carve((size_t)BT * Hh * 2);
  __bf16* VTb = (__bf16*)carve((size_t)BT * Hh * 2);
  __bf16* Ab  = (__bf16*)carve((size_t)BT * Hh * 2);
  __bf16* ATb = (__bf16*)carve((size_t)BT * Hh * 2);
  __bf16* Ob  = (__bf16*)carve((size_t)BT * Hh * 2);

  cvt_x_kernel<<<(BT * Dd + 255) / 256, 256, 0, stream>>>(x, xbf, BT * Dd);
  cvt_w_kernel<<<dim3((Dd * Hh + 255) / 256, 3), 256, 0, stream>>>(Wq, Wk, Wv, Wt);
  cvt_wo_kernel<<<(Dd * Hh + 255) / 256, 256, 0, stream>>>(Wo, Wot);

  qkv_kernel<<<dim3(BT / 16, 3), 32, 0, stream>>>(xbf, Wt, bq, bk, bv,
                                                  Qf, Kf, Qb, Vb, VTb);
  alpha_scan_kernel<<<Bb, Hh, 0, stream>>>(Qf, Kf, Ab, ATb);

  size_t attn_lds = 40960 + 4 * 16384;   // 104 KB dynamic LDS
  attn_kernel<<<Bb, 256, attn_lds, stream>>>(Qb, Vb, VTb, Ab, ATb, Ob);

  outproj_kernel<<<dim3(BT / 16, Dd / 32), 32, 0, stream>>>(Ob, Wot, bo, out);
}
